// Generator_Conv_62663572849140
// MI455X (gfx1250) — matrix operand order repaired and checked by reading, not compiled, not run on hardware
//
#include <hip/hip_runtime.h>
#include <hip/hip_bf16.h>
#include <math.h>

typedef __attribute__((ext_vector_type(16))) __bf16 v16bf;
typedef __attribute__((ext_vector_type(8)))  __bf16 v8bf;
typedef __attribute__((ext_vector_type(8)))  float  v8f;

#define BATCH   16
#define IN_C    512
#define OUT_C   256
#define HW      4096      // 64*64
#define H2      64

// ---------------------------------------------------------------------------
// Style affine: s[b][o] = SS * dot(style_base[b], aff_w[o]) + aff_b[o]
// ---------------------------------------------------------------------------
__global__ void style_affine_kernel(const float* __restrict__ sb,
                                    const float* __restrict__ w,
                                    const float* __restrict__ bvec,
                                    float* __restrict__ sout) {
    int idx = blockIdx.x * 256 + threadIdx.x;            // 16*512
    if (idx >= BATCH * 512) return;
    int b = idx >> 9, o = idx & 511;
    const float SS = 1.41421f / sqrtf(512.f);
    const float* sp = sb + b * 512;
    const float* wp = w + o * 512;
    float acc = 0.f;
    for (int k = 0; k < 512; ++k) acc = fmaf(sp[k], wp[k], acc);
    sout[idx] = SS * acc + bvec[o];
}

// ---------------------------------------------------------------------------
// Bilinear 2x upsample (align_corners) * WS1, fp32 -> bf16
// ---------------------------------------------------------------------------
__global__ void upsample_ws1_kernel(const float* __restrict__ cont,
                                    __bf16* __restrict__ up) {
    size_t idx = (size_t)blockIdx.x * 256 + threadIdx.x; // B*IN_C*4096
    if (idx >= (size_t)BATCH * IN_C * HW) return;
    const float WS1 = 1.41421f / sqrtf((float)(IN_C * 9));
    int xo = (int)(idx & 63), yo = (int)((idx >> 6) & 63);
    size_t bc = idx >> 12;
    float cy = yo * (31.f / 63.f);
    int y0 = (int)floorf(cy); float fy = cy - (float)y0;
    int y1 = y0 + 1; if (y1 > 31) y1 = 31;
    float cx = xo * (31.f / 63.f);
    int x0 = (int)floorf(cx); float fx = cx - (float)x0;
    int x1 = x0 + 1; if (x1 > 31) x1 = 31;
    const float* p = cont + bc * 1024;                   // 32*32
    float v00 = p[y0 * 32 + x0], v01 = p[y0 * 32 + x1];
    float v10 = p[y1 * 32 + x0], v11 = p[y1 * 32 + x1];
    float vh0 = v00 * (1.f - fy) + v10 * fy;
    float vh1 = v01 * (1.f - fy) + v11 * fy;
    up[idx] = (__bf16)((vh0 * (1.f - fx) + vh1 * fx) * WS1);
}

// ---------------------------------------------------------------------------
// Pack conv weights (OIHW fp32) into WMMA A-fragment-ready bf16 layout:
//   apack[((mt*KSTEPS + kstep)*32 + lane)*16 + e]
// K ordering: k = (r*3+s)*Cin + c  (each K=32 chunk has a single (r,s) tap).
// ISA 16-bit A 16x32 layout: lane<16 holds K {0..7,16..23},
// lane>=16 holds K {8..15,24..31}; elems 0..7 -> first run, 8..15 -> second.
// ---------------------------------------------------------------------------
__global__ void pack_weights_kernel(const float* __restrict__ w,
                                    __bf16* __restrict__ apack, int Cin) {
    int idx = blockIdx.x * 256 + threadIdx.x;
    int Ktot = Cin * 9;
    if (idx >= OUT_C * Ktot) return;
    int Ksteps = Ktot >> 5;
    int e    = idx & 15;
    int lane = (idx >> 4) & 31;
    int t    = idx >> 9;             // mt*Ksteps + kstep
    int kstep = t % Ksteps;
    int mt    = t / Ksteps;
    int half  = lane >> 4;
    int m     = mt * 16 + (lane & 15);
    int kin32 = (e < 8) ? (e + half * 8) : (e + 8 + half * 8);
    int k  = kstep * 32 + kin32;     // = rs*Cin + c
    int rs = k / Cin;
    int c  = k % Cin;
    int r = rs / 3, s = rs % 3;
    apack[idx] = (__bf16)w[((m * Cin + c) * 3 + r) * 3 + s];
}

// ---------------------------------------------------------------------------
// Implicit-GEMM conv3x3 via v_wmma_f32_16x16x32_bf16.
// Grid: 2048 blocks, one per 32-pixel N tile (32 contiguous x in one row).
// Block = 256 threads = 8 waves; wave w owns M-tiles {w, w+8} and both
// 16-pixel N subtiles -> 4 WMMAs per K-step per wave from 2 A loads.
// Double-buffered LDS B tile stored transposed [n][k]: fragment read is two
// 16B ds reads per lane; staging is 4 global u16 + 1 ds b64 store
// per thread per K-step.
// ---------------------------------------------------------------------------
template <int CIN>
__global__ __launch_bounds__(256) void conv_wmma_kernel(
    const __bf16* __restrict__ xin,   // (B, CIN, 64, 64) bf16
    const __bf16* __restrict__ apack, // packed weights
    const float*  __restrict__ bias,
    const float*  __restrict__ nscale,
    const float*  __restrict__ noise, // (B,1,64,64)
    const float*  __restrict__ alpha_p,
    float* __restrict__ out)          // (B, OUT_C, 64, 64)
{
    constexpr int KSTEPS = CIN * 9 / 32;
    constexpr int CSTEPS = CIN / 32;
    __shared__ __align__(64) __bf16 btile[2][32][32];   // [buf][n][k] 4KB

    const int tid  = threadIdx.x;
    const int lane = tid & 31;
    const int wave = tid >> 5;
    const int half = lane >> 4;
    const int col  = lane & 15;

    // Block-wide pixel tile: 32 contiguous x within one row of one image.
    const int n0 = blockIdx.x * 32;
    const int b  = n0 >> 12;
    const int y  = (n0 >> 6) & 63;
    const int x0 = n0 & 63;                  // 0 or 32

    // Staging role of this thread: pixel s_n (0..31), channel quad s_k.
    const int s_n = tid & 31;
    const int s_k = (tid >> 5) * 4;
    const int s_x = x0 + s_n;

    const __bf16* xb = xin + (size_t)b * CIN * HW;

    // A-fragment pointers for the two M-tiles of this wave.
    const __bf16* ap0 = apack + ((size_t)wave * KSTEPS) * 512 + (size_t)lane * 16;
    const __bf16* ap1 = ap0 + (size_t)8 * KSTEPS * 512;

    v8f acc00 = {}, acc01 = {}, acc10 = {}, acc11 = {};

    auto stage = [&](int step, int buf) {
        int rs = step / CSTEPS;
        int cc = (step - rs * CSTEPS) * 32;
        int r = rs / 3, s = rs - r * 3;
        int yy = y + r - 1, xx = s_x + s - 1;
        bool valid = (yy >= 0) & (yy < H2) & (xx >= 0) & (xx < H2);
        const __bf16* gp = xb + (size_t)(cc + s_k) * HW + (valid ? (yy * H2 + xx) : 0);
        #pragma unroll
        for (int j = 0; j < 4; ++j)          // adjacent stores -> merged ds b64
            btile[buf][s_n][s_k + j] = valid ? gp[(size_t)j * HW] : (__bf16)0.f;
    };

    stage(0, 0);
    __syncthreads();

    for (int step = 0; step < KSTEPS; ++step) {
        const int buf = step & 1;
        if (step + 1 < KSTEPS) stage(step + 1, buf ^ 1);

        // Prefetch A stream a few K-steps ahead (global_prefetch_b8).
        {
            int pf = (step + 4 < KSTEPS) ? (step + 4) : (KSTEPS - 1);
            __builtin_prefetch(ap0 + (size_t)pf * 512, 0, 1);
            __builtin_prefetch(ap1 + (size_t)pf * 512, 0, 1);
        }

        v16bf a0 = *(const v16bf*)(ap0 + (size_t)step * 512);
        v16bf a1 = *(const v16bf*)(ap1 + (size_t)step * 512);
        v16bf b0 = __builtin_shufflevector(*(const v8bf*)(&btile[buf][col][half * 8]), *(const v8bf*)(&btile[buf][col][16 + half * 8]), 0, 1, 2, 3, 4, 5, 6, 7, 8, 9, 10, 11, 12, 13, 14, 15);
        v16bf b1 = __builtin_shufflevector(*(const v8bf*)(&btile[buf][16 + col][half * 8]), *(const v8bf*)(&btile[buf][16 + col][16 + half * 8]), 0, 1, 2, 3, 4, 5, 6, 7, 8, 9, 10, 11, 12, 13, 14, 15);

        acc00 = __builtin_amdgcn_wmma_f32_16x16x32_bf16(
            false, a0, false, b0, (short)0, acc00, false, false);
        acc01 = __builtin_amdgcn_wmma_f32_16x16x32_bf16(
            false, a0, false, b1, (short)0, acc01, false, false);
        acc10 = __builtin_amdgcn_wmma_f32_16x16x32_bf16(
            false, a1, false, b0, (short)0, acc10, false, false);
        acc11 = __builtin_amdgcn_wmma_f32_16x16x32_bf16(
            false, a1, false, b1, (short)0, acc11, false, false);

        __syncthreads();
    }

    const float alpha = *alpha_p;
    const int px0 = y * H2 + x0 + col;       // N subtile 0 pixel
    const float nv0 = noise[(size_t)b * HW + px0];
    const float nv1 = noise[(size_t)b * HW + px0 + 16];
    #pragma unroll
    for (int v = 0; v < 8; ++v) {
        int m0 = wave * 16 + half * 8 + v;   // C/D layout: lane<16 -> M=v, lane>=16 -> M=8+v
        int m1 = m0 + 128;
        float bia0 = bias[m0], ns0 = nscale[m0];
        float bia1 = bias[m1], ns1v = nscale[m1];

        float t00 = acc00[v] + bia0 + ns0 * nv0;
        t00 = (t00 >= 0.f) ? t00 : alpha * t00;
        out[(size_t)(b * OUT_C + m0) * HW + px0] = t00;

        float t01 = acc01[v] + bia0 + ns0 * nv1;
        t01 = (t01 >= 0.f) ? t01 : alpha * t01;
        out[(size_t)(b * OUT_C + m0) * HW + px0 + 16] = t01;

        float t10 = acc10[v] + bia1 + ns1v * nv0;
        t10 = (t10 >= 0.f) ? t10 : alpha * t10;
        out[(size_t)(b * OUT_C + m1) * HW + px0] = t10;

        float t11 = acc11[v] + bia1 + ns1v * nv1;
        t11 = (t11 >= 0.f) ? t11 : alpha * t11;
        out[(size_t)(b * OUT_C + m1) * HW + px0 + 16] = t11;
    }
}

// ---------------------------------------------------------------------------
// Per-(b,c) mean / rstd (ddof=1) over 4096 pixels.
// ---------------------------------------------------------------------------
__global__ __launch_bounds__(256) void stats_kernel(const float* __restrict__ xin,
                                                    float2* __restrict__ st) {
    __shared__ float ssum[256];
    __shared__ float ssq[256];
    const int bc = blockIdx.x;                // b*256 + c
    const float* p = xin + (size_t)bc * HW;
    float s = 0.f, q = 0.f;
    for (int i = threadIdx.x; i < HW; i += 256) {
        float v = p[i];
        s += v; q = fmaf(v, v, q);
    }
    ssum[threadIdx.x] = s; ssq[threadIdx.x] = q;
    __syncthreads();
    for (int off = 128; off > 0; off >>= 1) {
        if ((int)threadIdx.x < off) {
            ssum[threadIdx.x] += ssum[threadIdx.x + off];
            ssq[threadIdx.x]  += ssq[threadIdx.x + off];
        }
        __syncthreads();
    }
    if (threadIdx.x == 0) {
        float mean = ssum[0] * (1.f / HW);
        float var  = (ssq[0] - ssum[0] * mean) * (1.f / (HW - 1));
        st[bc] = make_float2(mean, rsqrtf(var));
    }
}

// ---------------------------------------------------------------------------
// AdaIN1: normalize, apply style, * WS2, -> bf16 (input for conv2)
// ---------------------------------------------------------------------------
__global__ void adain_bf16_kernel(const float* __restrict__ xin,
                                  const float2* __restrict__ st,
                                  const float* __restrict__ sty,
                                  __bf16* __restrict__ out) {
    size_t idx = (size_t)blockIdx.x * 256 + threadIdx.x; // B*OUT_C*HW
    if (idx >= (size_t)BATCH * OUT_C * HW) return;
    const float WS2 = 1.41421f / sqrtf((float)(OUT_C * 9));
    int bc = (int)(idx >> 12);
    int b = bc >> 8, c = bc & 255;
    float2 mr = st[bc];
    float smean = sty[b * 512 + c];
    float sstd  = sty[b * 512 + 256 + c];
    float v = xin[idx];
    v = (sstd + 1.f) * ((v - mr.x) * mr.y) + smean;
    out[idx] = (__bf16)(v * WS2);
}

// ---------------------------------------------------------------------------
// AdaIN2: normalize, apply style -> fp32 final output
// ---------------------------------------------------------------------------
__global__ void adain_f32_kernel(const float* __restrict__ xin,
                                 const float2* __restrict__ st,
                                 const float* __restrict__ sty,
                                 float* __restrict__ out) {
    size_t idx = (size_t)blockIdx.x * 256 + threadIdx.x;
    if (idx >= (size_t)BATCH * OUT_C * HW) return;
    int bc = (int)(idx >> 12);
    int b = bc >> 8, c = bc & 255;
    float2 mr = st[bc];
    float smean = sty[b * 512 + c];
    float sstd  = sty[b * 512 + 256 + c];
    float v = xin[idx];
    out[idx] = (sstd + 1.f) * ((v - mr.x) * mr.y) + smean;
}

// ---------------------------------------------------------------------------
// Workspace layout (bytes)
// ---------------------------------------------------------------------------
#define UP_OFF   ((size_t)0)                    // 16*512*4096 bf16 = 64 MiB (reused as conv2 input)
#define X_OFF    ((size_t)67108864)             // 16*256*4096 fp32 = 64 MiB staging
#define A1_OFF   ((size_t)134217728)            // 256*4608 bf16
#define A2_OFF   ((size_t)136577024)            // 256*2304 bf16
#define S1_OFF   ((size_t)137756672)            // 16*512 fp32
#define S2_OFF   ((size_t)137789440)            // 16*512 fp32
#define ST_OFF   ((size_t)137822208)            // 16*256 float2

extern "C" void kernel_launch(void* const* d_in, const int* in_sizes, int n_in,
                              void* d_out, int out_size, void* d_ws, size_t ws_size,
                              hipStream_t stream) {
    const float* content   = (const float*)d_in[0];
    const float* style_base= (const float*)d_in[1];
    const float* noise     = (const float*)d_in[2];
    const float* conv1_w   = (const float*)d_in[3];
    const float* conv1_b   = (const float*)d_in[4];
    const float* conv2_w   = (const float*)d_in[5];
    const float* conv2_b   = (const float*)d_in[6];
    const float* aff1_w    = (const float*)d_in[7];
    const float* aff1_b    = (const float*)d_in[8];
    const float* aff2_w    = (const float*)d_in[9];
    const float* aff2_b    = (const float*)d_in[10];
    const float* prelu1_a  = (const float*)d_in[11];
    const float* prelu2_a  = (const float*)d_in[12];
    const float* ns1       = (const float*)d_in[13];
    const float* ns2       = (const float*)d_in[14];

    char* ws = (char*)d_ws;
    __bf16* up   = (__bf16*)(ws + UP_OFF);
    float*  xbuf = (float*)(ws + X_OFF);
    __bf16* a1   = (__bf16*)(ws + A1_OFF);
    __bf16* a2   = (__bf16*)(ws + A2_OFF);
    float*  s1   = (float*)(ws + S1_OFF);
    float*  s2   = (float*)(ws + S2_OFF);
    float2* st   = (float2*)(ws + ST_OFF);
    float*  outp = (float*)d_out;

    // styles (tiny GEMMs)
    style_affine_kernel<<<32, 256, 0, stream>>>(style_base, aff1_w, aff1_b, s1);
    style_affine_kernel<<<32, 256, 0, stream>>>(style_base, aff2_w, aff2_b, s2);

    // weight packing into WMMA fragment layout
    pack_weights_kernel<<<(OUT_C * IN_C * 9) / 256, 256, 0, stream>>>(conv1_w, a1, IN_C);
    pack_weights_kernel<<<(OUT_C * OUT_C * 9) / 256, 256, 0, stream>>>(conv2_w, a2, OUT_C);

    // upsample * WS1 -> bf16
    upsample_ws1_kernel<<<(BATCH * IN_C * HW) / 256, 256, 0, stream>>>(content, up);

    // conv1 (512 -> 256) + noise + PReLU
    conv_wmma_kernel<IN_C><<<BATCH * HW / 32, 256, 0, stream>>>(
        up, a1, conv1_b, ns1, noise, prelu1_a, xbuf);

    // AdaIN 1 (-> bf16, * WS2), reusing the upsample buffer
    stats_kernel<<<BATCH * OUT_C, 256, 0, stream>>>(xbuf, st);
    adain_bf16_kernel<<<(BATCH * OUT_C * HW) / 256, 256, 0, stream>>>(xbuf, st, s1, up);

    // conv2 (256 -> 256) + noise + PReLU
    conv_wmma_kernel<OUT_C><<<BATCH * HW / 32, 256, 0, stream>>>(
        up, a2, conv2_b, ns2, noise, prelu2_a, xbuf);

    // AdaIN 2 -> fp32 output
    stats_kernel<<<BATCH * OUT_C, 256, 0, stream>>>(xbuf, st);
    adain_f32_kernel<<<(BATCH * OUT_C * HW) / 256, 256, 0, stream>>>(xbuf, st, s2, outp);
}
